// BertAttention_83485574300235
// MI455X (gfx1250) — matrix operand order repaired and checked by reading, not compiled, not run on hardware
//
#include <hip/hip_runtime.h>
#include <hip/hip_bf16.h>

// ---------------------------------------------------------------------------
// BERT attention with previous-attention head-conv mixing, MI455X (gfx1250).
// bf16 WMMA (v_wmma_f32_16x16x32_bf16) for all GEMMs, fp32 softmax/LN.
// TDM (tensor_load_to_lds) stages K/V heads into LDS where available.
// ---------------------------------------------------------------------------

constexpr int   Bb    = 4;
constexpr int   Ss    = 1024;
constexpr int   Dd    = 768;
constexpr int   Hh    = 12;
constexpr int   DH    = 64;
constexpr int   NTOK  = Bb * Ss;       // 4096
constexpr float SCALE = 0.125f;        // 1/sqrt(64)
constexpr float ALPHA = 0.5f;
constexpr float EPS   = 1e-12f;

typedef __attribute__((ext_vector_type(16))) __bf16 bf16x16;
typedef __attribute__((ext_vector_type(8)))  float  floatx8;
typedef __attribute__((ext_vector_type(4)))  unsigned int u32x4;
typedef __attribute__((ext_vector_type(2)))  unsigned int u32x2;
typedef __attribute__((ext_vector_type(4)))  float  f32x4;
typedef __attribute__((ext_vector_type(8)))  int    i32x8;
typedef __attribute__((ext_vector_type(4)))  int    i32x4;

union ABOp { bf16x16 v; u32x4 u[2]; };

static __device__ __forceinline__ bf16x16 make_op(u32x4 lo, u32x4 hi) {
    ABOp t; t.u[0] = lo; t.u[1] = hi; return t.v;
}

static __device__ __forceinline__ floatx8 wmma_bf16(bf16x16 a, bf16x16 b, floatx8 c) {
    return __builtin_amdgcn_wmma_f32_16x16x32_bf16(false, a, false, b,
                                                   (short)0, c, false, false);
}

// LDS strides (bf16 elements unless noted).
constexpr int ASTR  = 40;     // 128x32 A tiles        (row =  80 B)
constexpr int BSTR  = 40;     // 64x32  B tiles        (row =  80 B)
constexpr int KSTR  = 72;     // 1024-row K head       (row = 144 B: 64 elem + 16 B pad)
constexpr int VROWE = 1040;   // 64-row V head         (row = 2080 B: 1024 elem + 2x16 B pad)
constexpr int KT_BYTES = 1024 * 144 + 16;   // 147472
constexpr int VT_BYTES = 64 * 2080;         // 133120

// ---------------------------------------------------------------------------
// Tensor Data Mover: 2-D tile load Global->LDS with LDS padding.
// D# layout per CDNA5 ISA ch.8 (group0: count/lds/global/type, group1: dims).
// ---------------------------------------------------------------------------
#if __has_builtin(__builtin_amdgcn_tensor_load_to_lds)
#define HAVE_TDM 1
static __device__ __forceinline__ void tdm_load_2d(unsigned lds_off, const void* gsrc,
        unsigned dim0, unsigned dim1, unsigned stride0,
        unsigned pad_en, unsigned pad_interval, unsigned pad_amount)
{
    unsigned long long ga = (unsigned long long)gsrc;
    u32x4 g0;
    g0.x = 1u;                                                // count=1, user D#
    g0.y = lds_off;                                           // lds_addr (bytes)
    g0.z = (unsigned)ga;                                      // global_addr[31:0]
    g0.w = (unsigned)((ga >> 32) & 0x1FFFFFFull) | (2u << 30); // addr[56:32] | type=2
    i32x8 g1;
    g1[0] = (int)((1u << 16) |                                // data_size = 2 B
                  (pad_en << 20) | (pad_interval << 22) | (pad_amount << 25));
    g1[1] = (int)((dim0 & 0xFFFFu) << 16);                    // tensor_dim0 lo16
    g1[2] = (int)((dim0 >> 16) | ((dim1 & 0xFFFFu) << 16));   // dim0 hi | dim1 lo
    g1[3] = (int)((dim1 >> 16) | ((dim0 & 0xFFFFu) << 16));   // dim1 hi | tile_dim0
    g1[4] = (int)(dim1 & 0xFFFFu);                            // tile_dim1 (tile==tensor)
    g1[5] = (int)stride0;                                     // tensor_dim0_stride lo32
    g1[6] = 0;
    g1[7] = 0;
    i32x4 z4 = {0, 0, 0, 0};
#if defined(__clang_major__) && (__clang_major__ >= 23)
    i32x8 z8 = {0, 0, 0, 0, 0, 0, 0, 0};
    __builtin_amdgcn_tensor_load_to_lds(g0, g1, z4, z4, z8, 0);
#else
    __builtin_amdgcn_tensor_load_to_lds(g0, g1, z4, z4, 0);
#endif
}
static __device__ __forceinline__ void tdm_wait0() {
#if __has_builtin(__builtin_amdgcn_s_wait_tensorcnt)
    __builtin_amdgcn_s_wait_tensorcnt(0);
#else
    asm volatile("s_wait_tensorcnt 0x0" ::: "memory");
#endif
}
#endif

// ---------------------------------------------------------------------------
// Kernel A: QKV projection.  X[4096,768]fp32 @ W[768,768]fp32 + b -> bf16.
// Q,K head-split [B,H,S,DH]; V stored transposed [B,H,DH,S] for k_ctx TDM.
// Double-buffered LDS tiles.  grid = (NTOK/128, H, 3), block = 256 (8 waves).
// ---------------------------------------------------------------------------
__global__ __launch_bounds__(256)
void k_qkv(const float* __restrict__ X,
           const float* __restrict__ Wq, const float* __restrict__ bq,
           const float* __restrict__ Wk, const float* __restrict__ bk,
           const float* __restrict__ Wv, const float* __restrict__ bv,
           __bf16* __restrict__ qb, __bf16* __restrict__ kbuf,
           __bf16* __restrict__ vbufT)
{
    __shared__ __bf16 At[2][128 * ASTR];   // 20 KB
    __shared__ __bf16 Bt[2][64 * BSTR];    // 10 KB

    const int t = threadIdx.x, wave = t >> 5, lane = t & 31;
    const int mbase = blockIdx.x * 128;
    const int h     = blockIdx.y;          // one 64-col tile == one head
    const int nbase = h * DH;
    const int which = blockIdx.z;

    const float* W    = which == 0 ? Wq : which == 1 ? Wk : Wv;
    const float* bias = which == 0 ? bq : which == 1 ? bk : bv;
    __bf16*      out  = which == 0 ? qb : which == 1 ? kbuf : vbufT;

    const int mrow = lane & 15;
    const int koff = (lane < 16) ? 0 : 8;
    const int bko  = (lane < 16) ? 0 : 8;
    const int mo   = (lane < 16) ? 0 : 8;

    auto stage = [&](int k0, int bsel) {
        #pragma unroll
        for (int i = 0; i < 4; ++i) {
            int e = (i * 256 + t) * 4;
            int r = e >> 5, c = e & 31;
            const f32x4 f = *(const f32x4*)&X[(mbase + r) * Dd + k0 + c];
            __bf16* dst = &At[bsel][r * ASTR + c];
            dst[0] = (__bf16)f.x; dst[1] = (__bf16)f.y;
            dst[2] = (__bf16)f.z; dst[3] = (__bf16)f.w;
        }
        int n = t & 63, kq = (t >> 6) * 8;
        #pragma unroll
        for (int kk = 0; kk < 8; ++kk)
            Bt[bsel][n * BSTR + kq + kk] = (__bf16)W[(k0 + kq + kk) * Dd + nbase + n];
    };

    floatx8 acc[4] = {};
    constexpr int NST = Dd / 32;           // 24 k-steps
    stage(0, 0);
    __syncthreads();
    for (int step = 0; step < NST; ++step) {
        const int cur = step & 1;
        if (step + 1 < NST) stage((step + 1) * 32, cur ^ 1);  // overlap next tile

        const char* arow = (const char*)&At[cur][(wave * 16 + mrow) * ASTR];
        bf16x16 a = make_op(*(const u32x4*)(arow + koff * 2),
                            *(const u32x4*)(arow + (koff + 16) * 2));
        #pragma unroll
        for (int ns = 0; ns < 4; ++ns) {
            const char* br = (const char*)&Bt[cur][(ns * 16 + mrow) * BSTR] + bko * 2;
            bf16x16 bm = make_op(*(const u32x4*)br, *(const u32x4*)(br + 32));
            acc[ns] = wmma_bf16(a, bm, acc[ns]);
        }
        __syncthreads();
    }

    #pragma unroll
    for (int ns = 0; ns < 4; ++ns) {
        int dh = ns * 16 + mrow;
        float bval = bias[nbase + dh];
        #pragma unroll
        for (int r = 0; r < 8; ++r) {
            int tok = mbase + wave * 16 + r + mo;
            int bi = tok >> 10, si = tok & 1023;
            long bh = (long)bi * Hh + h;
            long oidx = (which == 2) ? (bh * DH + dh) * Ss + si     // V transposed
                                     : (bh * Ss + si) * DH + dh;    // Q, K
            out[oidx] = (__bf16)(acc[ns][r] + bval);
        }
    }
}

// ---------------------------------------------------------------------------
// Kernel B: scores = Q.K^T * scale + mask -> HBM raw, flash-style per-row
// (max, sumexp) -> rowstats.  144 KB K head staged by TDM (fallback: copy).
// grid = (S/128, H, B), block = 256.
// ---------------------------------------------------------------------------
__global__ __launch_bounds__(256)
void k_scores(const __bf16* __restrict__ qb, const __bf16* __restrict__ kbuf,
              const float* __restrict__ mask,
              float* __restrict__ scores, float* __restrict__ rowstats)
{
    extern __shared__ char smem[];
    __bf16* Kt = (__bf16*)smem;                 // rows: 64 elem + 16 B pad (KSTR=72)

    const int t = threadIdx.x, wave = t >> 5, lane = t & 31;
    const int qblk = blockIdx.x * 128;
    const int h = blockIdx.y, bi = blockIdx.z;
    const long bh = (long)(bi * Hh + h);
    const __bf16* khead = kbuf + bh * Ss * DH;

#if defined(HAVE_TDM)
    if (wave == 0) {   // one wave issues the DMA; EXEC ignored by TDM
        // 2-D tile 64x1024 (2B elems), pad 4 dwords per 32-dword row -> stride 144 B
        tdm_load_2d((unsigned)(unsigned long long)(uintptr_t)Kt, khead,
                    /*dim0=*/64, /*dim1=*/1024, /*stride0=*/64,
                    /*pad_en=*/1, /*interval=*/4, /*amount=*/3);
        tdm_wait0();
    }
    __syncthreads();
#else
    const u32x4* src = (const u32x4*)khead;     // 8192 16-B vectors
    for (int i = 0; i < 32; ++i) {
        int j = i * 256 + t;
        int r = j >> 3, c8 = j & 7;
        *(u32x4*)&Kt[r * KSTR + c8 * 8] = src[j];
    }
    __syncthreads();
#endif

    const int mrow = lane & 15;
    const int koff = (lane < 16) ? 0 : 8;
    const int bko  = (lane < 16) ? 0 : 8;
    const int mo   = (lane < 16) ? 0 : 8;

    const __bf16* qrow = qb + (bh * Ss + (qblk + wave * 16 + mrow)) * DH;
    bf16x16 a0 = make_op(*(const u32x4*)((const char*)qrow + koff * 2),
                         *(const u32x4*)((const char*)qrow + (koff + 16) * 2));
    bf16x16 a1 = make_op(*(const u32x4*)((const char*)qrow + (32 + koff) * 2),
                         *(const u32x4*)((const char*)qrow + (32 + koff + 16) * 2));

    float mx[8], sm[8];
    #pragma unroll
    for (int r = 0; r < 8; ++r) { mx[r] = -3.0e38f; sm[r] = 0.f; }

    float* srow = scores + (bh * Ss + (qblk + wave * 16)) * (long)Ss;
    const float* mk_base = mask + bi * Ss;

    for (int nt = 0; nt < 64; ++nt) {
        floatx8 acc = {};
        const char* b0 = (const char*)&Kt[(nt * 16 + mrow) * KSTR] + bko * 2;
        acc = wmma_bf16(a0, make_op(*(const u32x4*)b0, *(const u32x4*)(b0 + 32)), acc);
        const char* b1 = (const char*)&Kt[(nt * 16 + mrow) * KSTR + 32] + bko * 2;
        acc = wmma_bf16(a1, make_op(*(const u32x4*)b1, *(const u32x4*)(b1 + 32)), acc);

        int n = nt * 16 + mrow;
        float mkv = mk_base[n];
        #pragma unroll
        for (int r = 0; r < 8; ++r) {
            float s = acc[r] * SCALE + mkv;
            srow[(r + mo) * (long)Ss + n] = s;
            float nm = fmaxf(mx[r], s);
            sm[r] = sm[r] * __expf(mx[r] - nm) + __expf(s - nm);
            mx[r] = nm;
        }
    }
    #pragma unroll
    for (int r = 0; r < 8; ++r) {
        float m = mx[r], s = sm[r];
        #pragma unroll
        for (int off = 1; off < 16; off <<= 1) {
            float om = __shfl_xor(m, off, 32);
            float os = __shfl_xor(s, off, 32);
            float nm = fmaxf(m, om);
            s = s * __expf(m - nm) + os * __expf(om - nm);
            m = nm;
        }
        if (mrow == 0) {
            int q = qblk + wave * 16 + r + mo;
            float* st = rowstats + (bh * Ss + q) * 2;
            st[0] = m; st[1] = s;
        }
    }
}

// ---------------------------------------------------------------------------
// Kernel C: head-conv of previous_attention + mix + second softmax -> bf16.
// grid = (S, B), block = 384 (wave o owns output head o).
// ---------------------------------------------------------------------------
__global__ __launch_bounds__(384)
void k_mix(const float* __restrict__ prevatt, const float* __restrict__ conv_w,
           const float* __restrict__ conv_b,
           const float* __restrict__ scores, const float* __restrict__ rowstats,
           __bf16* __restrict__ pmix)
{
    __shared__ float Pl[Hh * Ss];   // 48 KB: all 12 input-head rows at (b,q)
    const int t = threadIdx.x;
    const int q = blockIdx.x, bi = blockIdx.y;

    for (int i = 0; i < 32; ++i) {
        int idx = i * 384 + t;
        int ih = idx >> 10, k = idx & 1023;
        Pl[idx] = prevatt[(((long)(bi * Hh + ih)) * Ss + q) * Ss + k];
    }
    __syncthreads();

    const int o = t >> 5, lane = t & 31;
    float cw[Hh];
    #pragma unroll
    for (int i = 0; i < Hh; ++i) cw[i] = conv_w[o * Hh + i];
    const float cb = conv_b[o];

    const long row   = ((long)(bi * Hh + o) * Ss + q) * Ss;
    const float rmax = rowstats[((long)(bi * Hh + o) * Ss + q) * 2 + 0];
    const float rinv = 1.0f / rowstats[((long)(bi * Hh + o) * Ss + q) * 2 + 1];

    float lmax = -3.0e38f, lsum = 0.f;
    for (int kt = 0; kt < 32; ++kt) {
        int k = kt * 32 + lane;
        float pv = cb;
        #pragma unroll
        for (int i = 0; i < Hh; ++i) pv += cw[i] * Pl[i * Ss + k];
        float pb = __expf(scores[row + k] - rmax) * rinv;
        float m = ALPHA * pv + (1.f - ALPHA) * pb;
        float nm = fmaxf(lmax, m);
        lsum = lsum * __expf(lmax - nm) + __expf(m - nm);
        lmax = nm;
    }
    #pragma unroll
    for (int off = 1; off < 32; off <<= 1) {
        float om = __shfl_xor(lmax, off, 32);
        float os = __shfl_xor(lsum, off, 32);
        float nm = fmaxf(lmax, om);
        lsum = lsum * __expf(lmax - nm) + os * __expf(om - nm);
        lmax = nm;
    }
    const float inv = 1.0f / lsum;
    for (int kt = 0; kt < 32; ++kt) {           // recompute pass (Pl is hot)
        int k = kt * 32 + lane;
        float pv = cb;
        #pragma unroll
        for (int i = 0; i < Hh; ++i) pv += cw[i] * Pl[i * Ss + k];
        float pb = __expf(scores[row + k] - rmax) * rinv;
        float m = ALPHA * pv + (1.f - ALPHA) * pb;
        pmix[row + k] = (__bf16)(__expf(m - lmax) * inv);
    }
}

// ---------------------------------------------------------------------------
// Kernel D: ctx = probs @ V.  V head (already transposed [DH][S]) staged by
// TDM with padded rows (stride 2080 B); probs streamed as bf16 A operands.
// grid = (S/128, H, B), block = 256.
// ---------------------------------------------------------------------------
__global__ __launch_bounds__(256)
void k_ctx(const __bf16* __restrict__ pmix, const __bf16* __restrict__ vbufT,
           __bf16* __restrict__ ctxb)
{
    extern __shared__ char smem[];
    __bf16* Vt = (__bf16*)smem;                 // [64][VROWE] = 133120 B

    const int t = threadIdx.x, wave = t >> 5, lane = t & 31;
    const int qblk = blockIdx.x * 128;
    const int h = blockIdx.y, bi = blockIdx.z;
    const long bh = (long)(bi * Hh + h);
    const __bf16* vhead = vbufT + bh * DH * Ss;

#if defined(HAVE_TDM)
    if (wave == 0) {
        // 2-D tile 1024x64 (2B elems), pad 4 dwords per 256-dword interval
        tdm_load_2d((unsigned)(unsigned long long)(uintptr_t)Vt, vhead,
                    /*dim0=*/1024, /*dim1=*/64, /*stride0=*/1024,
                    /*pad_en=*/1, /*interval=*/7, /*amount=*/3);
        tdm_wait0();
    }
    __syncthreads();
#else
    for (int i = 0; i < 32; ++i) {              // 8192 16-B vectors
        int j = i * 256 + t;
        int n = j >> 7, kv = (j & 127) * 8;
        *(u32x4*)&Vt[n * VROWE + kv + ((kv >= 512) ? 8 : 0)] =
            ((const u32x4*)vhead)[j];
    }
    __syncthreads();
#endif

    const int mrow = lane & 15;
    const int koff = (lane < 16) ? 0 : 8;
    const int bko  = (lane < 16) ? 0 : 8;
    const int mo   = (lane < 16) ? 0 : 8;

    const __bf16* prow = pmix + (bh * Ss + (qblk + wave * 16 + mrow)) * (long)Ss;

    floatx8 acc[4] = {};
    for (int k0 = 0; k0 < Ss; k0 += 32) {
        if (k0 + 256 < Ss)
            __builtin_prefetch((const char*)prow + (k0 + 256) * 2, 0, 0);
        bf16x16 a = make_op(*(const u32x4*)((const char*)prow + (k0 + koff) * 2),
                            *(const u32x4*)((const char*)prow + (k0 + koff + 16) * 2));
        const int pad = (k0 >= 512) ? 16 : 0;   // mid-row TDM pad (bytes)
        #pragma unroll
        for (int ns = 0; ns < 4; ++ns) {
            const char* br = (const char*)Vt + (ns * 16 + mrow) * 2080
                             + (k0 + bko) * 2 + pad;
            bf16x16 bm = make_op(*(const u32x4*)br, *(const u32x4*)(br + 32));
            acc[ns] = wmma_bf16(a, bm, acc[ns]);
        }
    }

    #pragma unroll
    for (int ns = 0; ns < 4; ++ns) {
        int dh = ns * 16 + mrow;
        #pragma unroll
        for (int r = 0; r < 8; ++r) {
            int q = qblk + wave * 16 + r + mo;
            long tok = (long)bi * Ss + q;
            ctxb[tok * Dd + h * DH + dh] = (__bf16)acc[ns][r];
        }
    }
}

// ---------------------------------------------------------------------------
// Kernel E: out-proj: ctx(bf16) @ Wo + bo + hidden -> preln fp32.
// Double-buffered LDS tiles.  grid = (NTOK/128, D/64), block = 256.
// ---------------------------------------------------------------------------
__global__ __launch_bounds__(256)
void k_oproj(const __bf16* __restrict__ ctxb, const float* __restrict__ Wo,
             const float* __restrict__ bo, const float* __restrict__ hidden,
             float* __restrict__ preln)
{
    __shared__ __bf16 At[2][128 * ASTR];
    __shared__ __bf16 Bt[2][64 * BSTR];

    const int t = threadIdx.x, wave = t >> 5, lane = t & 31;
    const int mbase = blockIdx.x * 128;
    const int nbase = blockIdx.y * 64;

    const int mrow = lane & 15;
    const int koff = (lane < 16) ? 0 : 8;
    const int bko  = (lane < 16) ? 0 : 8;
    const int mo   = (lane < 16) ? 0 : 8;

    auto stage = [&](int k0, int bsel) {
        #pragma unroll
        for (int i = 0; i < 4; ++i) {
            int e = (i * 256 + t) * 4;
            int r = e >> 5, c = e & 31;
            *(u32x2*)&At[bsel][r * ASTR + c] =
                *(const u32x2*)&ctxb[(long)(mbase + r) * Dd + k0 + c];
        }
        int n = t & 63, kq = (t >> 6) * 8;
        #pragma unroll
        for (int kk = 0; kk < 8; ++kk)
            Bt[bsel][n * BSTR + kq + kk] = (__bf16)Wo[(k0 + kq + kk) * Dd + nbase + n];
    };

    floatx8 acc[4] = {};
    constexpr int NST = Dd / 32;
    stage(0, 0);
    __syncthreads();
    for (int step = 0; step < NST; ++step) {
        const int cur = step & 1;
        if (step + 1 < NST) stage((step + 1) * 32, cur ^ 1);

        const char* arow = (const char*)&At[cur][(wave * 16 + mrow) * ASTR];
        bf16x16 a = make_op(*(const u32x4*)(arow + koff * 2),
                            *(const u32x4*)(arow + (koff + 16) * 2));
        #pragma unroll
        for (int ns = 0; ns < 4; ++ns) {
            const char* br = (const char*)&Bt[cur][(ns * 16 + mrow) * BSTR] + bko * 2;
            bf16x16 bm = make_op(*(const u32x4*)br, *(const u32x4*)(br + 32));
            acc[ns] = wmma_bf16(a, bm, acc[ns]);
        }
        __syncthreads();
    }
    #pragma unroll
    for (int ns = 0; ns < 4; ++ns) {
        int c = nbase + ns * 16 + mrow;
        float bb = bo[c];
        #pragma unroll
        for (int r = 0; r < 8; ++r) {
            long tok = mbase + wave * 16 + r + mo;
            preln[tok * Dd + c] = acc[ns][r] + bb + hidden[tok * Dd + c];
        }
    }
}

// ---------------------------------------------------------------------------
// Kernel F: LayerNorm over D=768 per token.  grid = NTOK, block = 256.
// ---------------------------------------------------------------------------
__global__ __launch_bounds__(256)
void k_ln(const float* __restrict__ preln, const float* __restrict__ lnw,
          const float* __restrict__ lnb, float* __restrict__ out)
{
    __shared__ float red[2][8];
    const long tok = blockIdx.x;
    const int t = threadIdx.x;
    const float* row = preln + tok * Dd;
    float v[3], s = 0.f, s2 = 0.f;
    #pragma unroll
    for (int i = 0; i < 3; ++i) {
        v[i] = row[i * 256 + t];
        s += v[i]; s2 += v[i] * v[i];
    }
    #pragma unroll
    for (int off = 1; off < 32; off <<= 1) {
        s  += __shfl_xor(s,  off, 32);
        s2 += __shfl_xor(s2, off, 32);
    }
    const int wave = t >> 5, lane = t & 31;
    if (lane == 0) { red[0][wave] = s; red[1][wave] = s2; }
    __syncthreads();
    if (wave == 0) {
        float a  = (lane < 8) ? red[0][lane] : 0.f;
        float b2 = (lane < 8) ? red[1][lane] : 0.f;
        #pragma unroll
        for (int off = 1; off < 8; off <<= 1) {
            a  += __shfl_xor(a,  off, 32);
            b2 += __shfl_xor(b2, off, 32);
        }
        if (lane == 0) { red[0][0] = a; red[1][0] = b2; }
    }
    __syncthreads();
    const float mean = red[0][0] / Dd;
    const float var  = red[1][0] / Dd - mean * mean;
    const float rstd = rsqrtf(var + EPS);
    #pragma unroll
    for (int i = 0; i < 3; ++i) {
        int c = i * 256 + t;
        out[tok * Dd + c] = lnw[c] * ((v[i] - mean) * rstd) + lnb[c];
    }
}

// ---------------------------------------------------------------------------
extern "C" void kernel_launch(void* const* d_in, const int* in_sizes, int n_in,
                              void* d_out, int out_size, void* d_ws, size_t ws_size,
                              hipStream_t stream)
{
    (void)in_sizes; (void)n_in; (void)out_size; (void)ws_size;
    const float* hidden = (const float*)d_in[0];
    const float* mask   = (const float*)d_in[1];
    const float* prevat = (const float*)d_in[2];
    const float* Wq = (const float*)d_in[3];  const float* bq = (const float*)d_in[4];
    const float* Wk = (const float*)d_in[5];  const float* bk = (const float*)d_in[6];
    const float* Wv = (const float*)d_in[7];  const float* bv = (const float*)d_in[8];
    const float* cw = (const float*)d_in[9];  const float* cb = (const float*)d_in[10];
    const float* Wo = (const float*)d_in[11]; const float* bo = (const float*)d_in[12];
    const float* lnw = (const float*)d_in[13]; const float* lnb = (const float*)d_in[14];

    char* ws = (char*)d_ws;
    size_t off = 0;
    auto alloc = [&](size_t bytes) {
        char* p = ws + off;
        off = (off + bytes + 255) & ~(size_t)255;
        return p;
    };
    __bf16* qb       = (__bf16*)alloc((size_t)NTOK * Dd * 2);            //   6 MB
    __bf16* kbuf     = (__bf16*)alloc((size_t)NTOK * Dd * 2);            //   6 MB
    __bf16* vbufT    = (__bf16*)alloc((size_t)NTOK * Dd * 2);            //   6 MB
    float*  scores   = (float*) alloc((size_t)Bb * Hh * Ss * Ss * 4);    // 192 MB
    float*  rowstats = (float*) alloc((size_t)Bb * Hh * Ss * 2 * 4);     // 0.4 MB
    __bf16* pmix     = (__bf16*)alloc((size_t)Bb * Hh * Ss * Ss * 2);    //  96 MB
    __bf16* ctxb     = (__bf16*)alloc((size_t)NTOK * Dd * 2);            //   6 MB
    float*  preln    = (float*) alloc((size_t)NTOK * Dd * 4);            //  12 MB

    k_qkv   <<<dim3(NTOK / 128, Hh, 3), 256, 0, stream>>>(hidden, Wq, bq, Wk, bk, Wv, bv,
                                                          qb, kbuf, vbufT);
    k_scores<<<dim3(Ss / 128, Hh, Bb), 256, KT_BYTES, stream>>>(qb, kbuf, mask,
                                                                scores, rowstats);
    k_mix   <<<dim3(Ss, Bb), 384, 0, stream>>>(prevat, cw, cb, scores, rowstats, pmix);
    k_ctx   <<<dim3(Ss / 128, Hh, Bb), 256, VT_BYTES, stream>>>(pmix, vbufT, ctxb);
    k_oproj <<<dim3(NTOK / 128, Dd / 64), 256, 0, stream>>>(ctxb, Wo, bo, hidden, preln);
    k_ln    <<<NTOK, 256, 0, stream>>>(preln, lnw, lnb, (float*)d_out);
}
